// KANLSTMModel_46686294507754
// MI455X (gfx1250) — compile-verified, matrix operand order checked by reading, not run on hardware
//
#include <hip/hip_runtime.h>
#include <math.h>
#include <stdint.h>

#define BB 128
#define TT 1024
#define DD 512
#define HH 512
#define OUTD 256
#define NL 2
#define GKK 8
#define FOURH (4*HH)
#define KC 128   // K-chunk staged in LDS per iteration

typedef __attribute__((ext_vector_type(16))) __bf16 v16bf;
typedef __attribute__((ext_vector_type(8)))  __bf16 v8bf;
typedef __attribute__((ext_vector_type(8)))  float  v8f;

#if defined(__has_builtin)
#if __has_builtin(__builtin_amdgcn_s_wait_asynccnt)
#define USE_ASYNC_LDS 1
#endif
#endif
#ifndef USE_ASYNC_LDS
#define USE_ASYNC_LDS 0
#endif

__device__ __forceinline__ float sigm(float x) { return 1.0f / (1.0f + expf(-x)); }

// Cubic B-spline bases on the uniform extended grid: grid[j] = (j-3)*0.4 - 1, j=0..11
__device__ __forceinline__ void bspline8(float x, float* out) {
  const float hstep = 0.4f;
  float gr[12];
#pragma unroll
  for (int j = 0; j < 12; ++j) gr[j] = (float)(j - 3) * hstep - 1.0f;
  float bas[11];
#pragma unroll
  for (int j = 0; j < 11; ++j) bas[j] = (x >= gr[j] && x < gr[j + 1]) ? 1.0f : 0.0f;
#pragma unroll
  for (int k = 1; k <= 3; ++k) {
#pragma unroll
    for (int j = 0; j <= 10 - k; ++j) {
      float left  = (x - gr[j]) / (gr[j + k] - gr[j]);
      float right = (gr[j + k + 1] - x) / (gr[j + k + 1] - gr[j + 1]);
      bas[j] = left * bas[j] + right * bas[j + 1];
    }
  }
#pragma unroll
  for (int j = 0; j < 8; ++j) out[j] = bas[j];
}

// Cooperatively stage a 16 x KC bf16 weight panel (rows n0..n0+15) into LDS.
// 256 threads x 16 bytes each = 4 KB. Async global->LDS DMA when available.
__device__ __forceinline__ void stage_panel(__bf16* lds, const __bf16* g, int ld,
                                            int n0, int kc, int tid) {
  const int row = tid >> 4;          // 0..15
  const int col = (tid & 15) << 3;   // 0,8,...,120
  const __bf16* src = g + (size_t)(n0 + row) * ld + kc + col;
  __bf16* dst = lds + row * KC + col;
#if USE_ASYNC_LDS
  // vdst = LDS byte address (flat shared addr low 32 bits), vaddr = 64-bit global addr
  unsigned lds_addr = (unsigned)(uintptr_t)dst;
  unsigned long long gaddr = (unsigned long long)(uintptr_t)src;
  asm volatile("global_load_async_to_lds_b128 %0, %1, off"
               :: "v"(lds_addr), "v"(gaddr) : "memory");
#else
  *(v8bf*)dst = *(const v8bf*)src;
#endif
}

__device__ __forceinline__ void async_wait_all() {
#if USE_ASYNC_LDS
  __builtin_amdgcn_s_wait_asynccnt(0);
#endif
}

// Double-buffered GEMM pass: acc += A(m0..m0+15, :) * W(n0..n0+15, :)^T over K.
// Chunk k+1's panel DMA is issued before computing on chunk k, so the async
// global->LDS copy overlaps the WMMA stream.
// A_F32 = 1: A is f32 (converted in-flight); else A is bf16.
template <int A_F32>
__device__ __forceinline__ void gemm_pass_db(
    v8f& acc, const float* __restrict__ Af, const __bf16* __restrict__ Ab, int ldA,
    const __bf16* __restrict__ W, int ldW, int K,
    __bf16* lds0, __bf16* lds1,
    int m0, int n0, int arow, int kb16, int tid) {
  __bf16* bufs[2] = { lds0, lds1 };
  stage_panel(bufs[0], W, ldW, n0, 0, tid);
  async_wait_all();
  __syncthreads();
  int cur = 0;
  for (int kc = 0; kc < K; kc += KC) {
    if (kc + KC < K) stage_panel(bufs[cur ^ 1], W, ldW, n0, kc + KC, tid);  // prefetch next
    const __bf16* bp = bufs[cur];
#pragma unroll
    for (int kk = 0; kk < KC; kk += 32) {
      v16bf a;
      if (A_F32) {
        const float* p = Af + (size_t)(m0 + arow) * ldA + kc + kk + kb16;
#pragma unroll
        for (int j = 0; j < 16; ++j) a[j] = (__bf16)p[j];
      } else {
        a = *(const v16bf*)(Ab + (size_t)(m0 + arow) * ldA + kc + kk + kb16);
      }
      v16bf b = *(const v16bf*)(bp + arow * KC + kk + kb16);
      acc = __builtin_amdgcn_wmma_f32_16x16x32_bf16(false, a, false, b, (short)0, acc, false, false);
    }
    async_wait_all();    // next panel has landed (no-op on last chunk)
    __syncthreads();     // all waves done reading cur / next panel visible
    cur ^= 1;
  }
}

// ---------------- prep kernels ----------------
__global__ void cvt_kernel(__bf16* dst, const float* src, int n) {
  for (int i = blockIdx.x * blockDim.x + threadIdx.x; i < n; i += gridDim.x * blockDim.x)
    dst[i] = (__bf16)src[i];
}

__global__ void scale_cvt_kernel(__bf16* dst, const float* spline, const float* scaler, int n) {
  for (int i = blockIdx.x * blockDim.x + threadIdx.x; i < n; i += gridDim.x * blockDim.x)
    dst[i] = (__bf16)(spline[i] * scaler[i >> 3]);
}

__global__ void zero_state_kernel(float* hf, float* cf, __bf16* hb, int n) {
  for (int i = blockIdx.x * blockDim.x + threadIdx.x; i < n; i += gridDim.x * blockDim.x) {
    hf[i] = 0.0f; cf[i] = 0.0f; hb[i] = (__bf16)0.0f;
  }
}

// ---------------- gates GEMM + spline-feature epilogue ----------------
// gates = A @ Wih^T + h_prev @ Whh^T + bih + bhh
// A_F32=1: A = x[:,t,:] (f32, converted in-flight);  A_F32=0: A = h_bf of layer 0.
// epilogue: sigmoid(i,f,o) -> gates; for g: silu(g)->gb, bspline8(g)->sf (bf16 fragments).
template <int A_F32>
__global__ void __launch_bounds__(256)
gates_kernel_t(const float* __restrict__ xA, const __bf16* __restrict__ hA, int a_stride,
               const __bf16* __restrict__ Wih, const __bf16* __restrict__ Whh,
               const __bf16* __restrict__ hprev,
               const float* __restrict__ bih, const float* __restrict__ bhh,
               float* __restrict__ gates, __bf16* __restrict__ gb, __bf16* __restrict__ sf) {
  __shared__ alignas(64) __bf16 ldsB[2][16 * KC];
  const int tid  = threadIdx.x;
  const int widx = tid >> 5;        // wave in block -> M-tile
  const int lane = tid & 31;
  const int nt   = blockIdx.x;      // N-tile shared by all 8 waves
  const int m0 = widx * 16, n0 = nt * 16;
  const int arow = lane & 15;
  const int kb16 = (lane >> 4) * 16;
  v8f acc = {};
  // input contribution, K = 512
  gemm_pass_db<A_F32>(acc, xA, hA, a_stride, Wih, DD, DD,
                      ldsB[0], ldsB[1], m0, n0, arow, kb16, tid);
  // hidden contribution, K = 512
  gemm_pass_db<0>(acc, nullptr, hprev, HH, Whh, HH, HH,
                  ldsB[0], ldsB[1], m0, n0, arow, kb16, tid);

  const int ncol = n0 + (lane & 15);
  const float bias = bih[ncol] + bhh[ncol];
  const int gate = ncol >> 9;               // 0:i 1:f 2:g 3:o (uniform per tile)
  const int hcol = ncol & (HH - 1);
  const int rbase = m0 + ((lane >> 4) << 3);
#pragma unroll
  for (int r = 0; r < 8; ++r) {
    const int row = rbase + r;
    float v = acc[r] + bias;
    if (gate == 2) {
      gates[(size_t)row * FOURH + ncol] = v;                 // raw g
      gb[(size_t)row * HH + hcol] = (__bf16)(v * sigm(v));   // silu(g)
      float bas[8]; bspline8(v, bas);
      __bf16* sp = sf + (size_t)row * (HH * GKK) + (size_t)hcol * GKK;
#pragma unroll
      for (int j = 0; j < 8; ++j) sp[j] = (__bf16)bas[j];
    } else {
      gates[(size_t)row * FOURH + ncol] = sigm(v);
    }
  }
}

// ---------------- KAN GEMM + LSTM cell-update epilogue ----------------
// kan = silu(g) @ base^T + sf @ scaled^T ; c' = f*c + i*kan ; h' = o*tanh(c')
__global__ void __launch_bounds__(256)
kan_kernel(const __bf16* __restrict__ gb, const __bf16* __restrict__ sf,
           const __bf16* __restrict__ Wb, const __bf16* __restrict__ Ws,
           const float* __restrict__ gates,
           float* __restrict__ cf, float* __restrict__ hf, __bf16* __restrict__ hb) {
  __shared__ alignas(64) __bf16 ldsB[2][16 * KC];
  const int tid  = threadIdx.x;
  const int widx = tid >> 5;
  const int lane = tid & 31;
  const int nt   = blockIdx.x;      // 32 N-tiles (H=512)
  const int m0 = widx * 16, n0 = nt * 16;
  const int arow = lane & 15;
  const int kb16 = (lane >> 4) * 16;
  v8f acc = {};
  // base branch, K = 512
  gemm_pass_db<0>(acc, nullptr, gb, HH, Wb, HH, HH,
                  ldsB[0], ldsB[1], m0, n0, arow, kb16, tid);
  // spline branch, K = 4096
  gemm_pass_db<0>(acc, nullptr, sf, HH * GKK, Ws, HH * GKK, HH * GKK,
                  ldsB[0], ldsB[1], m0, n0, arow, kb16, tid);

  const int ncol = n0 + (lane & 15);
  const int rbase = m0 + ((lane >> 4) << 3);
#pragma unroll
  for (int r = 0; r < 8; ++r) {
    const int row = rbase + r;
    const float* grow = gates + (size_t)row * FOURH;
    float iv = grow[ncol];
    float fv = grow[512 + ncol];
    float ov = grow[1536 + ncol];
    float c  = cf[(size_t)row * HH + ncol];
    float cn = fv * c + iv * acc[r];
    float hn = ov * tanhf(cn);
    cf[(size_t)row * HH + ncol] = cn;
    hf[(size_t)row * HH + ncol] = hn;
    hb[(size_t)row * HH + ncol] = (__bf16)hn;
  }
}

// ---------------- final FC: out = h[1] @ fc_w^T + fc_b ----------------
__global__ void __launch_bounds__(256)
fc_kernel(const __bf16* __restrict__ hb, const float* __restrict__ W,
          const float* __restrict__ bias, float* __restrict__ out) {
  __shared__ alignas(64) __bf16 ldsB[16 * KC];
  const int tid  = threadIdx.x;
  const int widx = tid >> 5;
  const int lane = tid & 31;
  const int nt   = blockIdx.x;      // 16 N-tiles (O=256)
  const int m0 = widx * 16, n0 = nt * 16;
  const int arow = lane & 15;
  const int kb16 = (lane >> 4) * 16;
  v8f acc = {};
  for (int kc = 0; kc < HH; kc += KC) {
    { // stage f32 weights, converting to bf16 (no async: conversion needed)
      const int row = tid >> 4;
      const int col = (tid & 15) << 3;
      const float* p = W + (size_t)(n0 + row) * HH + kc + col;
      __bf16* dst = ldsB + row * KC + col;
#pragma unroll
      for (int j = 0; j < 8; ++j) dst[j] = (__bf16)p[j];
    }
    __syncthreads();
#pragma unroll
    for (int kk = 0; kk < KC; kk += 32) {
      v16bf a = *(const v16bf*)(hb + (size_t)(m0 + arow) * HH + kc + kk + kb16);
      v16bf b = *(const v16bf*)(ldsB + arow * KC + kk + kb16);
      acc = __builtin_amdgcn_wmma_f32_16x16x32_bf16(false, a, false, b, (short)0, acc, false, false);
    }
    __syncthreads();
  }
  const int ncol = n0 + (lane & 15);
  const int rbase = m0 + ((lane >> 4) << 3);
#pragma unroll
  for (int r = 0; r < 8; ++r)
    out[(size_t)(rbase + r) * OUTD + ncol] = acc[r] + bias[ncol];
}

extern "C" void kernel_launch(void* const* d_in, const int* in_sizes, int n_in,
                              void* d_out, int out_size, void* d_ws, size_t ws_size,
                              hipStream_t stream) {
  const float* x       = (const float*)d_in[0];
  const float* wih     = (const float*)d_in[1];
  const float* whh     = (const float*)d_in[2];
  const float* bih     = (const float*)d_in[3];
  const float* bhh     = (const float*)d_in[4];
  const float* kbase   = (const float*)d_in[5];
  const float* kspline = (const float*)d_in[6];
  const float* kscaler = (const float*)d_in[7];
  /* d_in[8] = grid (uniform, reconstructed analytically on device) */
  const float* fcw     = (const float*)d_in[9];
  const float* fcb     = (const float*)d_in[10];
  float* out = (float*)d_out;

  char* p = (char*)d_ws;
  auto alloc = [&](size_t bytes) -> void* {
    void* r = (void*)p;
    p += (bytes + 255) & ~(size_t)255;
    return r;
  };
  __bf16* wih_b    = (__bf16*)alloc(sizeof(__bf16) * (size_t)NL * FOURH * DD);
  __bf16* whh_b    = (__bf16*)alloc(sizeof(__bf16) * (size_t)NL * FOURH * HH);
  __bf16* base_b   = (__bf16*)alloc(sizeof(__bf16) * (size_t)NL * HH * HH);
  __bf16* scaled_b = (__bf16*)alloc(sizeof(__bf16) * (size_t)NL * HH * HH * GKK);
  __bf16* hb       = (__bf16*)alloc(sizeof(__bf16) * (size_t)NL * BB * HH);
  float*  hf       = (float*) alloc(sizeof(float)  * (size_t)NL * BB * HH);
  float*  cf       = (float*) alloc(sizeof(float)  * (size_t)NL * BB * HH);
  float*  gates    = (float*) alloc(sizeof(float)  * (size_t)BB * FOURH);
  __bf16* gbuf     = (__bf16*)alloc(sizeof(__bf16) * (size_t)BB * HH);
  __bf16* sfbuf    = (__bf16*)alloc(sizeof(__bf16) * (size_t)BB * HH * GKK);

  // one-time (per launch) weight conversion to bf16
  cvt_kernel<<<1024, 256, 0, stream>>>(wih_b, wih, NL * FOURH * DD);
  cvt_kernel<<<1024, 256, 0, stream>>>(whh_b, whh, NL * FOURH * HH);
  cvt_kernel<<<512, 256, 0, stream>>>(base_b, kbase, NL * HH * HH);
  scale_cvt_kernel<<<1024, 256, 0, stream>>>(scaled_b, kspline, kscaler, NL * HH * HH * GKK);
  zero_state_kernel<<<256, 256, 0, stream>>>(hf, cf, hb, NL * BB * HH);

  // sequential recurrence: 2 fused kernels per layer per timestep
  for (int t = 0; t < TT; ++t) {
    for (int l = 0; l < NL; ++l) {
      if (l == 0) {
        gates_kernel_t<1><<<128, 256, 0, stream>>>(
            x + (size_t)t * DD, nullptr, TT * DD,
            wih_b, whh_b, hb, bih, bhh, gates, gbuf, sfbuf);
      } else {
        gates_kernel_t<0><<<128, 256, 0, stream>>>(
            nullptr, hb, HH,
            wih_b + (size_t)FOURH * DD, whh_b + (size_t)FOURH * HH,
            hb + (size_t)BB * HH, bih + FOURH, bhh + FOURH, gates, gbuf, sfbuf);
      }
      kan_kernel<<<32, 256, 0, stream>>>(
          gbuf, sfbuf,
          base_b + (size_t)l * HH * HH, scaled_b + (size_t)l * HH * HH * GKK,
          gates,
          cf + (size_t)l * BB * HH, hf + (size_t)l * BB * HH, hb + (size_t)l * BB * HH);
    }
  }
  fc_kernel<<<16, 256, 0, stream>>>(hb + (size_t)BB * HH, fcw, fcb, out);
}
